// MultiHeadAttention_35897336660566
// MI455X (gfx1250) — compile-verified, matrix operand order checked by reading, not run on hardware
//
#include <hip/hip_runtime.h>
#include <hip/hip_bf16.h>

// ---------------------------------------------------------------------------
// MHA forward for MI455X (gfx1250): bf16 WMMA everywhere, flash-attention so
// the 512MB attn_bias is the only large HBM stream (scores never materialized).
// Workspace layout (bytes):
//   [0,8M)   Q   bf16 [B,H,T,64]
//   [8,16M)  K   bf16 [B,H,S,64]
//   [16,24M) V^T bf16 [B,H,64,S]
//   [24,32M) AO  bf16 [B,T,1024]   (pre-output-projection)
// ---------------------------------------------------------------------------

typedef __bf16 bf16;
typedef __attribute__((ext_vector_type(16))) __bf16 bf16x16;
typedef __attribute__((ext_vector_type(2)))  __bf16 bf16x2;
typedef __attribute__((ext_vector_type(8)))  float  f32x8;

#define D_MODEL 1024
#define NHEAD   16
#define HD      64
#define BATCH   2
#define SEQ     2048
#define ROWS    (BATCH * SEQ)   // 4096

// K index pattern for 16-bit A/B fragments (per ISA 7.12.2):
//   lanes 0-15  : kbase=0, lanes 16-31 : kbase=8
//   VGPR v<4 holds K = kbase+2v,2v+1 ; v>=4 holds K = kbase+16+2(v-4), +1
__device__ __forceinline__ int frag_kk(int v, int kb) {
  return kb + ((v < 4) ? (2 * v) : (16 + 2 * (v - 4)));
}

// Load a 16x32 A-frag (or B-frag of B^T) from an f32 row pointer, cvt->bf16.
// Pair addresses are contiguous, so the backend coalesces these into
// global_load_b128 + v_cvt_pk_bf16_f32 (verified in round-1 asm).
__device__ __forceinline__ bf16x16 load_frag_f32(const float* __restrict__ row, int kb) {
  bf16x16 f;
#pragma unroll
  for (int v = 0; v < 8; ++v) {
    const int kk = frag_kk(v, kb);
    const float2 p = *(const float2*)(row + kk);
    f[2 * v]     = (bf16)p.x;
    f[2 * v + 1] = (bf16)p.y;
  }
  return f;
}

// Same but from a bf16 row pointer (coalesces to two b128 loads per lane).
__device__ __forceinline__ bf16x16 load_frag_bf16(const bf16* __restrict__ row, int kb) {
  bf16x16 f;
#pragma unroll
  for (int v = 0; v < 8; ++v) {
    const int kk = frag_kk(v, kb);
    const bf16x2 p = *(const bf16x2*)(row + kk);
    f[2 * v]     = p[0];
    f[2 * v + 1] = p[1];
  }
  return f;
}

__device__ __forceinline__ f32x8 wmma_bf16(bf16x16 a, bf16x16 b, f32x8 c) {
  return __builtin_amdgcn_wmma_f32_16x16x32_bf16(false, a, false, b, (short)0, c,
                                                 false, false);
}

// ---------------------------------------------------------------------------
// Kernel 1: fused Q/K/V projection. blockIdx.z selects which projection.
// One wave -> 16 rows x 64 cols output strip; k-loop unrolled x2 so two
// independent load->cvt->wmma chains interleave under WMMA latency.
// ---------------------------------------------------------------------------
__global__ __launch_bounds__(256) void proj_qkv_kernel(
    const float* __restrict__ q_in, const float* __restrict__ k_in,
    const float* __restrict__ v_in,
    const float* __restrict__ Wq, const float* __restrict__ bq,
    const float* __restrict__ Wk, const float* __restrict__ bk,
    const float* __restrict__ Wv, const float* __restrict__ bv,
    bf16* __restrict__ Qws, bf16* __restrict__ Kws, bf16* __restrict__ Vtws)
{
  const int which = blockIdx.z;
  const float* X    = (which == 0) ? q_in : (which == 1) ? k_in : v_in;
  const float* W    = (which == 0) ? Wq   : (which == 1) ? Wk   : Wv;
  const float* bias = (which == 0) ? bq   : (which == 1) ? bk   : bv;

  const int lane = threadIdx.x & 31;
  const int gw   = blockIdx.x * 8 + (threadIdx.x >> 5);   // 4096 waves
  const int m0   = (gw >> 4) << 4;                        // 256 row tiles
  const int n0   = (gw & 15) << 6;                        // 16 col strips of 64
  const int ln   = lane & 15;                             // M for A, N for B
  const int kb   = (lane & 16) ? 8 : 0;
  const int mb   = (lane & 16) ? 8 : 0;                   // C/D row base

  const float* arow = X + (size_t)(m0 + ln) * D_MODEL;

  f32x8 acc[4] = {};
#pragma unroll 2
  for (int k0 = 0; k0 < D_MODEL; k0 += 32) {
    const bf16x16 a = load_frag_f32(arow + k0, kb);
#pragma unroll
    for (int nt = 0; nt < 4; ++nt) {
      // Y = X * W^T  =>  B[k][n] = W[n][k] : pairs contiguous along W row n
      const bf16x16 bf = load_frag_f32(W + (size_t)(n0 + nt * 16 + ln) * D_MODEL + k0, kb);
      acc[nt] = wmma_bf16(a, bf, acc[nt]);
    }
  }

#pragma unroll
  for (int nt = 0; nt < 4; ++nt) {
    const int col = n0 + nt * 16 + ln;
    const float bcol = bias[col];
    const int h = col >> 6, d = col & 63;
#pragma unroll
    for (int r = 0; r < 8; ++r) {
      const int row = m0 + mb + r;
      const float val = acc[nt][r] + bcol;
      const int b = row >> 11, t = row & 2047;
      if (which == 2) {        // V stored transposed: [B,H,64,S]
        Vtws[(((size_t)b * NHEAD + h) * HD + d) * SEQ + t] = (bf16)val;
      } else {                 // Q,K stored [B,H,S,64]
        bf16* dst = (which == 0) ? Qws : Kws;
        dst[(((size_t)b * NHEAD + h) * SEQ + t) * HD + d] = (bf16)val;
      }
    }
  }
}

// ---------------------------------------------------------------------------
// Kernel 2: flash attention. Block = 4 waves, one (b,h) x 16 query rows.
// Wave w owns SRC columns [w*512, w*512+512) in steps of 64: 4 score tiles,
// one online-softmax update, one LDS re-stripe, 8 AV WMMAs per step.
// ---------------------------------------------------------------------------
#define AW 4   // waves per block
__global__ __launch_bounds__(128) void attn_kernel(
    const bf16* __restrict__ Q, const bf16* __restrict__ K,
    const bf16* __restrict__ Vt, const float* __restrict__ bias,
    const unsigned char* __restrict__ kpm, bf16* __restrict__ AO)
{
  __shared__ bf16  p_lds[AW][16 * 64];   // P tile staging (C-layout -> A-layout)
  __shared__ float m_lds[AW][16];
  __shared__ float l_lds[AW][16];
  __shared__ float o_lds[AW][16][64];

  const int lane = threadIdx.x & 31;
  const int w    = threadIdx.x >> 5;
  const int t0   = blockIdx.x * 16;
  const int bh   = blockIdx.y;            // b*16 + h
  const int b    = bh >> 4;
  const int ln   = lane & 15;
  const int kb   = (lane & 16) ? 8 : 0;
  const int mb   = (lane & 16) ? 8 : 0;

  const size_t hoff = (size_t)bh * SEQ * HD;
  const bf16* Qh = Q  + hoff;
  const bf16* Kh = K  + hoff;
  const bf16* Vh = Vt + hoff;                       // [64][SEQ]
  const float* bh_bias = bias + (size_t)bh * SEQ * SEQ;
  const float NEG_INF = -__builtin_inff();

  // Q A-fragments (head_dim 64 -> two k=32 chunks), kept in registers.
  const bf16* qrow = Qh + (size_t)(t0 + ln) * HD;
  const bf16x16 qa0 = load_frag_bf16(qrow, kb);
  const bf16x16 qa1 = load_frag_bf16(qrow + 32, kb);

  float mrow[8], lrow[8];
  f32x8 oacc[4] = {};
#pragma unroll
  for (int r = 0; r < 8; ++r) { mrow[r] = NEG_INF; lrow[r] = 0.f; }

  const float scale = 0.125f;   // 1/sqrt(64)
  const int s_end = w * 512 + 512;

  for (int s0 = w * 512; s0 < s_end; s0 += 64) {
    // prefetch next bias block (speculative; OOB silently dropped)
    __builtin_prefetch(bh_bias + (size_t)(t0 + mb) * SEQ + s0 + 64 + ln, 0, 0);

    // ---- scores: four 16x16 tiles over columns [s0, s0+64) ----
    float sc[4][8];
#pragma unroll
    for (int j = 0; j < 4; ++j) {
      const bf16* krow = Kh + (size_t)(s0 + 16 * j + ln) * HD;  // B[k][n]=K[s][k]
      f32x8 a = {};
      a = wmma_bf16(qa0, load_frag_bf16(krow, kb), a);
      a = wmma_bf16(qa1, load_frag_bf16(krow + 32, kb), a);
      const int scol = s0 + 16 * j + ln;
      const bool masked = kpm[b * SEQ + scol] != 0;
      const float* bcol = bh_bias + (size_t)(t0 + mb) * SEQ + scol;
#pragma unroll
      for (int r = 0; r < 8; ++r) {
        const float v = a[r] * scale + bcol[(size_t)r * SEQ];
        sc[j][r] = masked ? NEG_INF : v;
      }
    }

    // ---- online softmax row statistics ----
    float mnew[8], corr[8];
#pragma unroll
    for (int r = 0; r < 8; ++r) {
      float mx = fmaxf(fmaxf(sc[0][r], sc[1][r]), fmaxf(sc[2][r], sc[3][r]));
#pragma unroll
      for (int off = 8; off; off >>= 1)   // reduce across the 16-lane half
        mx = fmaxf(mx, __shfl_xor(mx, off, 32));
      const float mn = fmaxf(mrow[r], mx);
      corr[r] = (mrow[r] > NEG_INF) ? __expf(mrow[r] - mn) : 0.f;
      mnew[r] = mn;
    }

    // ---- P = exp(s - m); stage to LDS in natural row-major for A-frag reload
    float psum[8] = {0, 0, 0, 0, 0, 0, 0, 0};
    bf16* pl = &p_lds[w][0];
#pragma unroll
    for (int j = 0; j < 4; ++j) {
      const int col = 16 * j + ln;
#pragma unroll
      for (int r = 0; r < 8; ++r) {
        const float p = (mnew[r] > NEG_INF) ? __expf(sc[j][r] - mnew[r]) : 0.f;
        pl[(mb + r) * 64 + col] = (bf16)p;
        psum[r] += p;
      }
    }
#pragma unroll
    for (int r = 0; r < 8; ++r) {
#pragma unroll
      for (int off = 8; off; off >>= 1)
        psum[r] += __shfl_xor(psum[r], off, 32);
      lrow[r] = lrow[r] * corr[r] + psum[r];
      mrow[r] = mnew[r];
    }
#pragma unroll
    for (int nt = 0; nt < 4; ++nt)
#pragma unroll
      for (int r = 0; r < 8; ++r)
        oacc[nt][r] *= corr[r];

    // same-wave LDS RAW: drain DS counter before re-striping P as A-fragments
    asm volatile("s_wait_dscnt 0" ::: "memory");

    const bf16x16 pa0 = load_frag_bf16(pl + ln * 64, kb);        // P k=0..31
    const bf16x16 pa1 = load_frag_bf16(pl + ln * 64 + 32, kb);   // P k=32..63
#pragma unroll
    for (int nt = 0; nt < 4; ++nt) {
      // B[k][n] = V[s0+k][nt*16+n] = Vt[nt*16+n][s0+k] (contiguous pairs)
      const bf16* vrow = Vh + (size_t)(nt * 16 + ln) * SEQ + s0;
      oacc[nt] = wmma_bf16(pa0, load_frag_bf16(vrow, kb), oacc[nt]);
      oacc[nt] = wmma_bf16(pa1, load_frag_bf16(vrow + 32, kb), oacc[nt]);
    }
  }

  // ---- combine the 4 per-wave partials ----
  if (ln == 0) {
#pragma unroll
    for (int r = 0; r < 8; ++r) {
      m_lds[w][mb + r] = mrow[r];
      l_lds[w][mb + r] = lrow[r];
    }
  }
#pragma unroll
  for (int nt = 0; nt < 4; ++nt)
#pragma unroll
    for (int r = 0; r < 8; ++r)
      o_lds[w][mb + r][nt * 16 + ln] = oacc[nt][r];

  __syncthreads();

  const int tid  = threadIdx.x;
  const int row  = tid >> 3;          // 0..15
  const int dblk = (tid & 7) * 8;     // 0..56
  float M = NEG_INF;
#pragma unroll
  for (int ww = 0; ww < AW; ++ww) M = fmaxf(M, m_lds[ww][row]);
  float e[AW], L = 0.f;
#pragma unroll
  for (int ww = 0; ww < AW; ++ww) {
    e[ww] = (m_lds[ww][row] > NEG_INF) ? __expf(m_lds[ww][row] - M) : 0.f;
    L += l_lds[ww][row] * e[ww];
  }
  const float invL = (L > 0.f) ? 1.f / L : 0.f;
  const size_t obase = ((size_t)b * SEQ + (t0 + row)) * D_MODEL + (bh & 15) * HD;
#pragma unroll
  for (int d = 0; d < 8; ++d) {
    float acc = 0.f;
#pragma unroll
    for (int ww = 0; ww < AW; ++ww) acc += o_lds[ww][row][dblk + d] * e[ww];
    AO[obase + dblk + d] = (bf16)(acc * invL);
  }
}

// ---------------------------------------------------------------------------
// Kernel 3: output projection (bf16 A from workspace, f32 weights, f32 out).
// ---------------------------------------------------------------------------
__global__ __launch_bounds__(256) void out_proj_kernel(
    const bf16* __restrict__ AOws, const float* __restrict__ Wo,
    const float* __restrict__ bo, float* __restrict__ out)
{
  const int lane = threadIdx.x & 31;
  const int gw   = blockIdx.x * 8 + (threadIdx.x >> 5);
  const int m0   = (gw >> 4) << 4;
  const int n0   = (gw & 15) << 6;
  const int ln   = lane & 15;
  const int kb   = (lane & 16) ? 8 : 0;
  const int mb   = (lane & 16) ? 8 : 0;

  const bf16* arow = AOws + (size_t)(m0 + ln) * D_MODEL;

  f32x8 acc[4] = {};
#pragma unroll 2
  for (int k0 = 0; k0 < D_MODEL; k0 += 32) {
    const bf16x16 a = load_frag_bf16(arow + k0, kb);
#pragma unroll
    for (int nt = 0; nt < 4; ++nt) {
      const bf16x16 bf = load_frag_f32(Wo + (size_t)(n0 + nt * 16 + ln) * D_MODEL + k0, kb);
      acc[nt] = wmma_bf16(a, bf, acc[nt]);
    }
  }
#pragma unroll
  for (int nt = 0; nt < 4; ++nt) {
    const int col = n0 + nt * 16 + ln;
    const float bcol = bo[col];
#pragma unroll
    for (int r = 0; r < 8; ++r)
      out[(size_t)(m0 + mb + r) * D_MODEL + col] = acc[nt][r] + bcol;
  }
}

// ---------------------------------------------------------------------------
extern "C" void kernel_launch(void* const* d_in, const int* in_sizes, int n_in,
                              void* d_out, int out_size, void* d_ws, size_t ws_size,
                              hipStream_t stream) {
  const float* q   = (const float*)d_in[0];
  const float* k   = (const float*)d_in[1];
  const float* v   = (const float*)d_in[2];
  const float* ab  = (const float*)d_in[3];
  const unsigned char* kpm = (const unsigned char*)d_in[4];  // bool mask
  const float* Wq  = (const float*)d_in[5];
  const float* bq  = (const float*)d_in[6];
  const float* Wk  = (const float*)d_in[7];
  const float* bk  = (const float*)d_in[8];
  const float* Wv  = (const float*)d_in[9];
  const float* bv  = (const float*)d_in[10];
  const float* Wo  = (const float*)d_in[11];
  const float* bo  = (const float*)d_in[12];
  float* out = (float*)d_out;

  char* ws = (char*)d_ws;
  const size_t MB = 1024 * 1024;
  bf16* Qws = (bf16*)(ws);
  bf16* Kws = (bf16*)(ws + 8 * MB);
  bf16* Vt  = (bf16*)(ws + 16 * MB);
  bf16* AO  = (bf16*)(ws + 24 * MB);

  // 1) Q/K/V projections: 4096 waves each, z selects projection.
  proj_qkv_kernel<<<dim3(512, 1, 3), 256, 0, stream>>>(
      q, k, v, Wq, bq, Wk, bk, Wv, bv, Qws, Kws, Vt);

  // 2) Flash attention: grid (T/16, B*H), 4 waves per block.
  attn_kernel<<<dim3(SEQ / 16, BATCH * NHEAD), 128, 0, stream>>>(
      Qws, Kws, Vt, ab, kpm, AO);

  // 3) Output projection to f32.
  out_proj_kernel<<<dim3(512), 256, 0, stream>>>(AO, Wo, bo, out);
}
